// LSTM_38869454028951
// MI455X (gfx1250) — matrix operand order repaired and checked by reading, not compiled, not run on hardware
//
#include <hip/hip_runtime.h>

typedef __attribute__((ext_vector_type(16))) _Float16 v16h;
typedef __attribute__((ext_vector_type(8)))  _Float16 v8h;
typedef __attribute__((ext_vector_type(8)))  float    v8f;

#define B_  512
#define T_  365
#define F_  16
#define H_  256

#define SPH 264   // layer-1 h buffer row stride (f16 elems), padded for LDS banks
#define SPX 32    // layer-1 x buffer row stride
#define SP2 520   // layer-2 [x2|h] buffer row stride

#if __has_builtin(__builtin_amdgcn_global_load_async_to_lds_b128) && \
    __has_builtin(__builtin_amdgcn_s_wait_asynccnt)
#define USE_ASYNC_LDS 1
// The builtin takes pointers to 16-byte int vectors in AS1 (global) / AS3 (LDS).
typedef int v4i_vs __attribute__((__vector_size__(16)));
typedef __attribute__((address_space(1))) v4i_vs* async_gptr_t;
typedef __attribute__((address_space(3))) v4i_vs* async_lptr_t;
#endif

#if __has_builtin(__builtin_amdgcn_sched_barrier)
#define SCHED_PIN() __builtin_amdgcn_sched_barrier(0)
#else
#define SCHED_PIN()
#endif

__device__ __forceinline__ float sigf(float x)       { return 1.0f / (1.0f + __expf(-x)); }
__device__ __forceinline__ float tanh_fast(float x)  { return 2.0f / (1.0f + __expf(-2.0f * x)) - 1.0f; }

__device__ __forceinline__ v16h cat16(v8h lo, v8h hi) {
  return __builtin_shufflevector(lo, hi, 0,1,2,3,4,5,6,7,8,9,10,11,12,13,14,15);
}
// A fragment: elements 0..7 = K+0..7, elements 8..15 = K+16..23 (per-lane base)
__device__ __forceinline__ v16h loadA(const _Float16* p) {
  return cat16(*(const v8h*)p, *(const v8h*)(p + 16));
}
// B fragment: elements 0..7 = K+0..7, elements 8..15 = K+16..23 (per-lane base)
__device__ __forceinline__ v16h loadB(const _Float16* p) {
  return cat16(*(const v8h*)p, *(const v8h*)(p + 16));
}
#define WMMA(Am, Bm, Cm) \
  __builtin_amdgcn_wmma_f32_16x16x32_f16(false, (Am), false, (Bm), (short)0, (Cm), false, false)

// ---------------------------------------------------------------------------
// Prep: convert weights to f16 in WMMA-friendly layouts, combine biases.
//   wih1p column 16 carries the combined layer-1 bias (paired with a constant
//   1.0 in xbuf column 16), so layer-1 accumulators can start from zero.
// ---------------------------------------------------------------------------
__global__ void lstm_prep_kernel(const float* __restrict__ Wih1, const float* __restrict__ Whh1,
                                 const float* __restrict__ bih1, const float* __restrict__ bhh1,
                                 const float* __restrict__ Wih2, const float* __restrict__ Whh2,
                                 const float* __restrict__ bih2, const float* __restrict__ bhh2,
                                 _Float16* whh1h, _Float16* wih1p, _Float16* wcat2,
                                 float* b1c, float* b2c) {
  int i = blockIdx.x * blockDim.x + threadIdx.x;
  if (i < 1024 * 256) { whh1h[i] = (_Float16)Whh1[i]; return; }
  i -= 1024 * 256;
  if (i < 1024 * 32) {
    int n = i >> 5, f = i & 31;
    _Float16 v;
    if (f < 16)       v = (_Float16)Wih1[n * 16 + f];
    else if (f == 16) v = (_Float16)(bih1[n] + bhh1[n]);   // folded bias
    else              v = (_Float16)0.0f;
    wih1p[i] = v;
    return;
  }
  i -= 1024 * 32;
  if (i < 1024 * 512) {
    int n = i >> 9, k = i & 511;
    wcat2[i] = (k < 256) ? (_Float16)Wih2[n * 256 + k]
                         : (_Float16)Whh2[n * 256 + (k - 256)];
    return;
  }
  i -= 1024 * 512;
  if (i < 1024) { b1c[i] = bih1[i] + bhh1[i]; return; }
  i -= 1024;
  if (i < 1024) { b2c[i] = bih2[i] + bhh2[i]; return; }
}

// ---------------------------------------------------------------------------
// Layer 1: each workgroup owns 16 batch rows for all T steps; 16 waves, wave w
// owns hidden units j0=w*16..+15 across all 4 gates. z^T = W x h^T:
//   A (16x32 f16) = weight rows from global (L2-resident), 2-deep pipelined
//   B (32x16 f16) = h^T from LDS
//   D (16x16 f32) rows = gate-output idx, cols = batch -> wave-local update.
// Bias enters via wih1p col 16 x xbuf col 16 (==1.0): no bias registers.
// ---------------------------------------------------------------------------
__global__ __launch_bounds__(512)
void lstm_layer1_kernel(const float* __restrict__ x,
                        const _Float16* whh1h, const _Float16* wih1p,
                        _Float16* h1seq) {
  __shared__ _Float16 hbuf[16 * SPH];
  __shared__ _Float16 xbuf[16 * SPX];

  const int tid  = threadIdx.x;
  const int wave = tid >> 5;
  const int lane = tid & 31;
  const int lh   = lane & 15;
  const bool hi  = lane >= 16;
  const int m0   = blockIdx.x * 16;
  const int j0   = wave * 16;
  const int rbase = hi ? 8 : 0;
  const int ofsA  = hi ? 8 : 0;
  const int ofsB  = hi ? 8 : 0;

  for (int i = tid; i < 16 * SPH; i += 512) hbuf[i] = (_Float16)0.0f;
  for (int i = tid; i < 16 * SPX; i += 512) xbuf[i] = (_Float16)0.0f;
  if (tid < 16) xbuf[tid * SPX + 16] = (_Float16)1.0f;   // bias multiplier col

  // Input-projection A fragments (K=32, col16 = bias): constant over t.
  v16h aih[4];
  #pragma unroll
  for (int g = 0; g < 4; ++g)
    aih[g] = loadA(wih1p + (size_t)(g * 256 + j0 + lh) * 32 + ofsA);

  const _Float16* wrow[4];
  #pragma unroll
  for (int g = 0; g < 4; ++g)
    wrow[g] = whh1h + (size_t)(g * 256 + j0 + lh) * 256 + ofsA;

  v8f c = {};
  #pragma unroll 1
  for (int t = 0; t < T_; ++t) {
    if (tid < 256) {
      int m = tid >> 4, f = tid & 15;
      xbuf[m * SPX + f] = (_Float16)x[((size_t)(m0 + m) * T_ + t) * F_ + f];
    }
    __syncthreads();

    v8f a0 = {}, a1 = {}, a2 = {}, a3 = {};

    // Pipeline prologue: issue A/B loads for chunks 0 and 1, pinned before
    // the x-contribution WMMAs so they overlap.
    v16h Af[2][4];
    v16h Bf[2];
    #pragma unroll
    for (int g = 0; g < 4; ++g) Af[0][g] = loadA(wrow[g] + 0);
    #pragma unroll
    for (int g = 0; g < 4; ++g) Af[1][g] = loadA(wrow[g] + 32);
    Bf[0] = loadB(&hbuf[lh * SPH + 0 + ofsB]);
    Bf[1] = loadB(&hbuf[lh * SPH + 32 + ofsB]);
    SCHED_PIN();

    // x + bias contribution (preloaded A; overlaps prologue load latency).
    {
      v16h bx = loadB(&xbuf[lh * SPX + ofsB]);
      a0 = WMMA(aih[0], bx, a0);
      a1 = WMMA(aih[1], bx, a1);
      a2 = WMMA(aih[2], bx, a2);
      a3 = WMMA(aih[3], bx, a3);
    }

    // Recurrent K=256, 8 chunks, 2-deep software pipeline held in place by
    // sched_barrier: refill loads for chunk kc+2 stay ahead of WMMAs(kc).
    #pragma unroll
    for (int kc = 0; kc < 8; ++kc) {
      const int cur = kc & 1;
      v16h bh = Bf[cur];
      v16h f0 = Af[cur][0], f1 = Af[cur][1], f2 = Af[cur][2], f3 = Af[cur][3];
      if (kc + 2 < 8) {
        const int kn = (kc + 2) * 32;
        #pragma unroll
        for (int g = 0; g < 4; ++g) Af[cur][g] = loadA(wrow[g] + kn);
        Bf[cur] = loadB(&hbuf[lh * SPH + kn + ofsB]);
      }
      SCHED_PIN();
      a0 = WMMA(f0, bh, a0);
      a1 = WMMA(f1, bh, a1);
      a2 = WMMA(f2, bh, a2);
      a3 = WMMA(f3, bh, a3);
    }

    v8h hv;
    #pragma unroll
    for (int r = 0; r < 8; ++r) {
      float iv = sigf(a0[r]);
      float fv = sigf(a1[r]);
      float gv = tanh_fast(a2[r]);
      float ov = sigf(a3[r]);
      float cv = fv * c[r] + iv * gv;
      c[r] = cv;
      hv[r] = (_Float16)(ov * tanh_fast(cv));
    }

    __syncthreads();  // all hbuf reads complete before overwrite
    *(v8h*)&hbuf[lh * SPH + j0 + rbase] = hv;
    *(v8h*)&h1seq[((size_t)(m0 + lh) * T_ + t) * H_ + j0 + rbase] = hv;
  }
}

// ---------------------------------------------------------------------------
// Layer 2 + linear head: A = [W_ih2 | W_hh2] (K=512), B = [h1_t ; h]^T in one
// LDS buffer. h1_t staged via gfx1250 async global->LDS when available.
// ---------------------------------------------------------------------------
__global__ __launch_bounds__(512)
void lstm_layer2_kernel(const _Float16* h1seq, const _Float16* wcat2,
                        const float* b2c,
                        const float* __restrict__ Wlin, const float* __restrict__ blin,
                        float* __restrict__ out) {
  __shared__ _Float16 xh[16 * SP2];

  const int tid  = threadIdx.x;
  const int wave = tid >> 5;
  const int lane = tid & 31;
  const int lh   = lane & 15;
  const bool hi  = lane >= 16;
  const int m0   = blockIdx.x * 16;
  const int j0   = wave * 16;
  const int rbase = hi ? 8 : 0;
  const int ofsA  = hi ? 8 : 0;
  const int ofsB  = hi ? 8 : 0;

  for (int i = tid; i < 16 * SP2; i += 512) xh[i] = (_Float16)0.0f;

  v8f bias0, bias1, bias2, bias3;
  #pragma unroll
  for (int r = 0; r < 8; ++r) {
    bias0[r] = b2c[0 * 256 + j0 + r + rbase];
    bias1[r] = b2c[1 * 256 + j0 + r + rbase];
    bias2[r] = b2c[2 * 256 + j0 + r + rbase];
    bias3[r] = b2c[3 * 256 + j0 + r + rbase];
  }
  const _Float16* wrow[4];
  #pragma unroll
  for (int g = 0; g < 4; ++g)
    wrow[g] = wcat2 + (size_t)(g * 256 + j0 + lh) * 512 + ofsA;

  v8f c = {};
  #pragma unroll 1
  for (int t = 0; t < T_; ++t) {
    // Stage h1_t tile (16 x 256 f16) into xh cols 0..255: 16B per thread.
    {
      int m = tid >> 5, seg = tid & 31;
      const _Float16* gp = h1seq + ((size_t)(m0 + m) * T_ + t) * H_ + seg * 8;
      _Float16* lp = &xh[m * SP2 + seg * 8];
#ifdef USE_ASYNC_LDS
      __builtin_amdgcn_global_load_async_to_lds_b128(
          (async_gptr_t)gp, (async_lptr_t)lp, 0, 0);
      __builtin_amdgcn_s_wait_asynccnt(0);
#else
      *(v8h*)lp = *(const v8h*)gp;
#endif
    }
    __syncthreads();

    v8f a0 = bias0, a1 = bias1, a2 = bias2, a3 = bias3;

    v16h Af[2][4];
    v16h Bf[2];
    #pragma unroll
    for (int g = 0; g < 4; ++g) Af[0][g] = loadA(wrow[g] + 0);
    #pragma unroll
    for (int g = 0; g < 4; ++g) Af[1][g] = loadA(wrow[g] + 32);
    Bf[0] = loadB(&xh[lh * SP2 + 0 + ofsB]);
    Bf[1] = loadB(&xh[lh * SP2 + 32 + ofsB]);
    SCHED_PIN();

    // K=512 in 16 chunks, 2-deep software pipeline with sched_barrier pins.
    #pragma unroll
    for (int kc = 0; kc < 16; ++kc) {
      const int cur = kc & 1;
      v16h bh = Bf[cur];
      v16h f0 = Af[cur][0], f1 = Af[cur][1], f2 = Af[cur][2], f3 = Af[cur][3];
      if (kc + 2 < 16) {
        const int kn = (kc + 2) * 32;
        #pragma unroll
        for (int g = 0; g < 4; ++g) Af[cur][g] = loadA(wrow[g] + kn);
        Bf[cur] = loadB(&xh[lh * SP2 + kn + ofsB]);
      }
      SCHED_PIN();
      a0 = WMMA(f0, bh, a0);
      a1 = WMMA(f1, bh, a1);
      a2 = WMMA(f2, bh, a2);
      a3 = WMMA(f3, bh, a3);
    }

    v8h hv;
    #pragma unroll
    for (int r = 0; r < 8; ++r) {
      float iv = sigf(a0[r]);
      float fv = sigf(a1[r]);
      float gv = tanh_fast(a2[r]);
      float ov = sigf(a3[r]);
      float cv = fv * c[r] + iv * gv;
      c[r] = cv;
      hv[r] = (_Float16)(ov * tanh_fast(cv));
    }

    __syncthreads();
    *(v8h*)&xh[lh * SP2 + 256 + j0 + rbase] = hv;   // h region = cols 256..511
  }

  __syncthreads();
  if (tid < 16) {
    float s = blin[0];
    const _Float16* hrow = &xh[tid * SP2 + 256];
    for (int j = 0; j < 256; ++j) s += (float)hrow[j] * Wlin[j];
    out[m0 + tid] = s;
  }
}

// ---------------------------------------------------------------------------
// Workspace layout (bytes):
//   [0)        whh1h : 524288      [524288)  wih1p : 65536
//   [589824)   wcat2 : 1048576     [1638400) b1c   : 4096
//   [1642496)  b2c   : 4096        [1646592) h1seq : 95682560   (~97.3 MB)
// ---------------------------------------------------------------------------
extern "C" void kernel_launch(void* const* d_in, const int* in_sizes, int n_in,
                              void* d_out, int out_size, void* d_ws, size_t ws_size,
                              hipStream_t stream) {
  const float* x     = (const float*)d_in[0];
  const float* Wih1  = (const float*)d_in[1];
  const float* Whh1  = (const float*)d_in[2];
  const float* bih1  = (const float*)d_in[3];
  const float* bhh1  = (const float*)d_in[4];
  const float* Wih2  = (const float*)d_in[5];
  const float* Whh2  = (const float*)d_in[6];
  const float* bih2  = (const float*)d_in[7];
  const float* bhh2  = (const float*)d_in[8];
  const float* Wlin  = (const float*)d_in[9];
  const float* blin  = (const float*)d_in[10];
  float* out = (float*)d_out;

  char* ws = (char*)d_ws;
  _Float16* whh1h = (_Float16*)(ws + 0);
  _Float16* wih1p = (_Float16*)(ws + 524288);
  _Float16* wcat2 = (_Float16*)(ws + 589824);
  float*    b1c   = (float*)   (ws + 1638400);
  float*    b2c   = (float*)   (ws + 1642496);
  _Float16* h1seq = (_Float16*)(ws + 1646592);

  lstm_prep_kernel<<<821248 / 256, 256, 0, stream>>>(
      Wih1, Whh1, bih1, bhh1, Wih2, Whh2, bih2, bhh2,
      whh1h, wih1p, wcat2, b1c, b2c);

  lstm_layer1_kernel<<<B_ / 16, 512, 0, stream>>>(x, whh1h, wih1p, h1seq);

  lstm_layer2_kernel<<<B_ / 16, 512, 0, stream>>>(h1seq, wcat2, b2c, Wlin, blin, out);
}
